// HyperbolicAttention_56573309223547
// MI455X (gfx1250) — matrix operand order repaired and checked by reading, not compiled, not run on hardware
//
#include <hip/hip_runtime.h>
#include <hip/hip_bf16.h>
#include <math.h>

typedef __attribute__((ext_vector_type(16))) __bf16 v16bf;
typedef __attribute__((ext_vector_type(8)))  __bf16 v8bf;
typedef __attribute__((ext_vector_type(8)))  float  v8f;

#define LDSW 136  // padded LDS row stride (bf16 elems): 272 B rotates banks by 4/row

// ---------------------------------------------------------------------------
// out[n, c] = sum_k X[n,k] * W[c,k] + b[c]     (X: [nrows,128], W: [128,128])
// 256 threads = 8 waves; each wave computes a 16x128 strip via
// v_wmma_f32_16x16x32_bf16 (4 K-steps x 8 column tiles).
// ---------------------------------------------------------------------------
__global__ __launch_bounds__(256)
void gemm_xwt_kernel(const float* __restrict__ X, const float* __restrict__ W,
                     const float* __restrict__ bias, float* __restrict__ out,
                     int nrows)
{
    __shared__ __bf16 Wl[128 * LDSW];
    __shared__ float  bl[128];
    const int tid = threadIdx.x;

    // Stage W[c][k] into LDS as bf16 (vectorized f32x4 loads)
    for (int i = tid; i < 128 * 32; i += 256) {
        const int r  = i >> 5;
        const int c4 = (i & 31) << 2;
        const float4 w = *(const float4*)(W + r * 128 + c4);
        __bf16* dst = &Wl[r * LDSW + c4];
        dst[0] = (__bf16)w.x; dst[1] = (__bf16)w.y;
        dst[2] = (__bf16)w.z; dst[3] = (__bf16)w.w;
    }
    if (tid < 128) bl[tid] = bias[tid];
    __syncthreads();

    const int wave = tid >> 5;
    const int lane = tid & 31;
    const int half = lane >> 4;   // half-wave select
    const int l16  = lane & 15;
    const int row0 = blockIdx.x * 128 + wave * 16;
    if (row0 >= nrows) return;    // wave-uniform: EXEC stays all-ones for WMMA

    v8f acc[8] = {};

    for (int kk = 0; kk < 128; kk += 32) {
        // A fragment: lane holds row M=l16; elems 0..7 -> K=kk+half*8+e,
        // elems 8..15 -> K=kk+16+half*8+e  (16-bit A 16x32 layout, ISA 7.12.2)
        int arow = row0 + l16;
        if (arow >= nrows) arow = nrows - 1;   // clamp keeps EXEC full
        const float* ap = X + arow * 128 + kk + half * 8;
        const v8f a0 = *(const v8f*)ap;
        const v8f a1 = *(const v8f*)(ap + 16);
        v16bf A;
#pragma unroll
        for (int e = 0; e < 8; ++e) { A[e] = (__bf16)a0[e]; A[8 + e] = (__bf16)a1[e]; }

#pragma unroll
        for (int ct = 0; ct < 8; ++ct) {
            // B fragment: lane holds column N=ct*16+l16 of W^T, i.e. row c of W;
            // elems 0..7 -> K = kk + half*8 + e, elems 8..15 -> K = kk + 16 + half*8 + e
            const __bf16* bp = &Wl[(ct * 16 + l16) * LDSW + kk + half * 8];
            union { v16bf v; struct { v8bf lo, hi; } p; } B;
            B.p.lo = *(const v8bf*)bp;
            B.p.hi = *(const v8bf*)(bp + 16);
            acc[ct] = __builtin_amdgcn_wmma_f32_16x16x32_bf16(
                false, A, false, B.v, (short)0, acc[ct], false, false);
        }
    }

    // D layout: lane -> col l16, VGPR r -> row r + half*8
#pragma unroll
    for (int ct = 0; ct < 8; ++ct) {
        const float b = bl[ct * 16 + l16];
#pragma unroll
        for (int r = 0; r < 8; ++r) {
            const int m = row0 + r + half * 8;
            if (m < nrows) out[m * 128 + ct * 16 + l16] = acc[ct][r] + b;
        }
    }
}

// ---- order-preserving float <-> uint encoding for atomic max ---------------
__device__ __forceinline__ unsigned encode_max(float f) {
    unsigned b = __float_as_uint(f);
    return (b & 0x80000000u) ? ~b : (b | 0x80000000u);
}
__device__ __forceinline__ float decode_max(unsigned u) {
    unsigned b = (u & 0x80000000u) ? (u ^ 0x80000000u) : ~u;
    return __uint_as_float(b);
}

// ---------------------------------------------------------------------------
// scores[e*8+h] = dot(q[row[e],h,:], k[col[e],h,:]) / 4 ; per-head global max
// h == tid & 7 is invariant (stride multiple of 8), so block reduction can
// keep the mod-8 classes separated.
// ---------------------------------------------------------------------------
__global__ __launch_bounds__(256)
void scores_kernel(const float* __restrict__ q, const float* __restrict__ k,
                   const int* __restrict__ rows, const int* __restrict__ cols,
                   float* __restrict__ scores, unsigned* __restrict__ hmax, int E)
{
    __shared__ float red[256];
    const int tid = threadIdx.x;
    const int stride = gridDim.x * 256;
    float lmax = -INFINITY;
    for (int i = blockIdx.x * 256 + tid; i < E * 8; i += stride) {
        const int e = i >> 3, h = i & 7;
        const float4* qp = (const float4*)(q + rows[e] * 128 + h * 16);
        const float4* kp = (const float4*)(k + cols[e] * 128 + h * 16);
        float s = 0.f;
#pragma unroll
        for (int j = 0; j < 4; ++j) {
            const float4 a = qp[j], b = kp[j];
            s += a.x * b.x + a.y * b.y + a.z * b.z + a.w * b.w;
        }
        s *= 0.25f;                 // 1/sqrt(D), D=16
        scores[i] = s;
        lmax = fmaxf(lmax, s);
    }
    red[tid] = lmax;
    __syncthreads();
    for (int st = 128; st >= 8; st >>= 1) {
        if (tid < st) red[tid] = fmaxf(red[tid], red[tid + st]);
        __syncthreads();
    }
    if (tid < 8) atomicMax(&hmax[tid], encode_max(red[tid]));
}

// scores[i] <- exp(scores[i] - max[h]); hsum[h] += sum of exps
__global__ __launch_bounds__(256)
void expsum_kernel(float* __restrict__ scores, const unsigned* __restrict__ hmax,
                   float* __restrict__ hsum, int E)
{
    __shared__ float red[256];
    const int tid = threadIdx.x;
    const float mx = decode_max(hmax[tid & 7]);
    const int stride = gridDim.x * 256;
    float lsum = 0.f;
    for (int i = blockIdx.x * 256 + tid; i < E * 8; i += stride) {
        const float ex = __expf(scores[i] - mx);
        scores[i] = ex;
        lsum += ex;
    }
    red[tid] = lsum;
    __syncthreads();
    for (int st = 128; st >= 8; st >>= 1) {
        if (tid < st) red[tid] += red[tid + st];
        __syncthreads();
    }
    if (tid < 8) atomicAdd(&hsum[tid], red[tid]);
}

// accum[row[e], c] += (exp_score[e,h]/sum[h]) * v[col[e], c]   (h = c/16)
__global__ __launch_bounds__(256)
void message_kernel(const float* __restrict__ v, const float* __restrict__ scores,
                    const float* __restrict__ hsum,
                    const int* __restrict__ rows, const int* __restrict__ cols,
                    float* __restrict__ accum, int E)
{
    float rs[8];
#pragma unroll
    for (int h = 0; h < 8; ++h) rs[h] = 1.0f / hsum[h];
    const int total  = E * 128;             // 81.92M, fits int32
    const int stride = gridDim.x * 256;
    for (int i = blockIdx.x * 256 + threadIdx.x; i < total; i += stride) {
        const int e = i >> 7, c = i & 127, h = c >> 4;
        const float w = scores[e * 8 + h] * rs[h];
        atomicAdd(&accum[rows[e] * 128 + c], w * v[cols[e] * 128 + c]);
    }
}

// ---------------------------------------------------------------------------
extern "C" void kernel_launch(void* const* d_in, const int* in_sizes, int n_in,
                              void* d_out, int out_size, void* d_ws, size_t ws_size,
                              hipStream_t stream)
{
    const float* features = (const float*)d_in[0];
    const int*   edge     = (const int*)d_in[1];   // [2, E] int32
    const float* Wq = (const float*)d_in[2]; const float* bq = (const float*)d_in[3];
    const float* Wk = (const float*)d_in[4]; const float* bk = (const float*)d_in[5];
    const float* Wv = (const float*)d_in[6]; const float* bv = (const float*)d_in[7];
    const float* Wo = (const float*)d_in[8]; const float* bo = (const float*)d_in[9];
    float* out = (float*)d_out;

    const int N = in_sizes[0] / 128;
    const int E = in_sizes[1] / 2;
    const int* rows = edge;        // edge_index[0]
    const int* cols = edge + E;    // edge_index[1]

    const size_t nodeBytes = (size_t)N * 128 * sizeof(float);
    char* ws = (char*)d_ws;
    size_t off = 0;
    float* q      = (float*)(ws + off); off += nodeBytes;
    float* kbuf   = (float*)(ws + off); off += nodeBytes;
    float* vbuf   = (float*)(ws + off); off += nodeBytes;
    float* accum  = (float*)(ws + off); off += nodeBytes;
    float* scores = (float*)(ws + off); off += (size_t)E * 8 * sizeof(float);
    unsigned* hmax = (unsigned*)(ws + off); off += 64;   // 8 u32, padded
    float*    hsum = (float*)(ws + off);    off += 64;   // 8 f32, padded

    hipMemsetAsync(accum, 0, nodeBytes, stream);
    hipMemsetAsync(hmax, 0, 128, stream);   // clears hmax (encode floor) + hsum (0.0f)

    const dim3 blk(256);
    const int gemmGrid = (N + 127) / 128;
    gemm_xwt_kernel<<<gemmGrid, blk, 0, stream>>>(features, Wq, bq, q,    N);
    gemm_xwt_kernel<<<gemmGrid, blk, 0, stream>>>(features, Wk, bk, kbuf, N);
    gemm_xwt_kernel<<<gemmGrid, blk, 0, stream>>>(features, Wv, bv, vbuf, N);

    const int ehGrid = (E * 8 + 255) / 256;
    scores_kernel<<<ehGrid, blk, 0, stream>>>(q, kbuf, rows, cols, scores, hmax, E);
    expsum_kernel<<<ehGrid, blk, 0, stream>>>(scores, hmax, hsum, E);

    const int msgGrid = (E * 128 + 255) / 256;
    message_kernel<<<msgGrid, blk, 0, stream>>>(vbuf, scores, hsum, rows, cols, accum, E);

    gemm_xwt_kernel<<<gemmGrid, blk, 0, stream>>>(accum, Wo, bo, out, N);
}
